// RNN_model_83107617177620
// MI455X (gfx1250) — compile-verified, matrix operand order checked by reading, not run on hardware
//
#include <hip/hip_runtime.h>
#include <hip/hip_bf16.h>

typedef __attribute__((ext_vector_type(16))) __bf16 v16bf;
typedef __attribute__((ext_vector_type(8)))  __bf16 v8bf;
typedef __attribute__((ext_vector_type(8)))  float  v8f;
typedef __attribute__((ext_vector_type(4)))  unsigned int v4u;
typedef __attribute__((ext_vector_type(4)))  int v4i;
typedef __attribute__((ext_vector_type(8)))  int v8i;

#define DEVFN __device__ __forceinline__

constexpr int B_  = 128;
constexpr int T_  = 256;
constexpr int E_  = 300;
constexpr int H_  = 512;
constexpr int G4  = 4 * H_;   // 2048 gate columns (i, j, f, o)
constexpr int KX  = 320;      // E padded to multiple of 32
constexpr int NREC = 6;       // 3 sequences x 2 directions
constexpr int WGS_PER_REC = 16;
constexpr int NCOLS = 32;     // hidden columns per WG
constexpr int LDS_BYTES = 4 * NCOLS * H_ * 2;   // 128 KB Wh slice

DEVFN unsigned short f2bf(float f) {
  unsigned u = __float_as_uint(f);
  unsigned r = u + 0x7fffu + ((u >> 16) & 1u);   // round-to-nearest-even
  return (unsigned short)(r >> 16);
}
DEVFN float bf2f(unsigned short b) { return __uint_as_float(((unsigned)b) << 16); }

// One wave32 WMMA 16-bit operand fragment for this lane:
// two contiguous 16-byte chunks, 32 bytes apart (K-chunk striping per ISA layout).
DEVFN v16bf load_frag(const unsigned short* p) {
  v8bf lo = *reinterpret_cast<const v8bf*>(p);
  v8bf hi = *reinterpret_cast<const v8bf*>(p + 16);
  v16bf r;
#pragma unroll
  for (int i = 0; i < 8; ++i) { r[i] = lo[i]; r[i + 8] = hi[i]; }
  return r;
}

DEVFN float sigm(float x) { return 1.0f / (1.0f + __expf(-x)); }
DEVFN float ftanh(float x) {       // branch-free tanh: pure VALU transcendental
  x = fminf(fmaxf(x, -15.0f), 15.0f);
  float e = __expf(2.0f * x);
  return (e - 1.0f) / (e + 1.0f);
}

// ---------------- pre-processing kernels ----------------

__global__ void k_convert_x(const float* __restrict__ q, const float* __restrict__ a,
                            const float* __restrict__ an, unsigned short* __restrict__ X) {
  long idx = (long)blockIdx.x * blockDim.x + threadIdx.x;
  const long total = (long)3 * T_ * B_ * KX;
  if (idx >= total) return;
  int k = (int)(idx % KX); long r = idx / KX;
  int b = (int)(r % B_);   r /= B_;
  int t = (int)(r % T_);   int seq = (int)(r / T_);
  const float* xs = (seq == 0) ? q : ((seq == 1) ? a : an);
  float v = (k < E_) ? xs[((long)b * T_ + t) * E_ + k] : 0.0f;   // input layout [B,T,E]
  X[idx] = f2bf(v);
}

__global__ void k_build_wxT(const float* __restrict__ Wf, const float* __restrict__ Wb,
                            unsigned short* __restrict__ WxT) {
  long idx = (long)blockIdx.x * blockDim.x + threadIdx.x;
  const long total = (long)2 * G4 * KX;
  if (idx >= total) return;
  int k = (int)(idx % KX); long r = idx / KX;
  int n = (int)(r % G4);   int d = (int)(r / G4);
  const float* W = d ? Wb : Wf;
  float v = (k < E_) ? W[(size_t)k * G4 + n] : 0.0f;
  WxT[idx] = f2bf(v);
}

__global__ void k_build_whT(const float* __restrict__ Wf, const float* __restrict__ Wb,
                            unsigned short* __restrict__ WhT) {
  long idx = (long)blockIdx.x * blockDim.x + threadIdx.x;
  const long total = (long)2 * G4 * H_;
  if (idx >= total) return;
  int k = (int)(idx % H_); long r = idx / H_;
  int n = (int)(r % G4);   int d = (int)(r / G4);
  const float* W = d ? Wb : Wf;
  WhT[idx] = f2bf(W[(size_t)(E_ + k) * G4 + n]);
}

// ---------------- main recurrent WMMA kernel ----------------
// grid = NREC * WGS_PER_REC blocks, 256 threads (8 waves of 32).
// Each wave owns a 16-row batch tile; each WG owns a 32-wide hidden slice whose
// Wh columns (128 KB bf16) are DMA'd into LDS once by the Tensor Data Mover.
// Cell state lives in VGPRs across all T steps.

__global__ void __launch_bounds__(256, 1)
k_lstm(const unsigned short* __restrict__ X,
       const unsigned short* __restrict__ WxT,
       const unsigned short* __restrict__ WhT,
       const float* __restrict__ bfw, const float* __restrict__ bbw,
       unsigned short* __restrict__ Hbuf,    // [2][NREC][128][512] double-buffered h
       unsigned short* __restrict__ Hout,    // [3][2][T][128][512] history (bf16)
       unsigned int* __restrict__ syncc) {
  extern __shared__ unsigned short sh_wh[];  // [4 gates][NCOLS][512] bf16

  const int rec  = blockIdx.x / WGS_PER_REC;
  const int wg   = blockIdx.x % WGS_PER_REC;
  const int seq  = rec >> 1;
  const int dir  = rec & 1;
  const int wave = threadIdx.x >> 5;
  const int lane = threadIdx.x & 31;
  const int lr   = lane & 15;
  const int kh   = (lane >> 4) & 1;     // K-chunk / M-half selector per WMMA layout
  const int m0   = wave * 16;           // batch-row tile base
  const int n0   = wg * NCOLS;          // hidden-column slice base

  const float* bias = dir ? bbw : bfw;
  const unsigned short* WxTd = WxT + (size_t)dir * G4 * KX;
  const unsigned short* WhTd = WhT + (size_t)dir * G4 * H_;
  const size_t hstride = (size_t)NREC * B_ * H_;

  // ---- Stage this WG's Wh slice into LDS via the Tensor Data Mover ----
#if __has_builtin(__builtin_amdgcn_tensor_load_to_lds)
  if (wave == 0) {
    const unsigned lds_base = (unsigned)(uintptr_t)(void*)sh_wh;  // low 32b = LDS offset
    const unsigned chunkB = NCOLS * H_ * 2;          // bytes per gate block (32 KB)
    const unsigned chunkE = chunkB / 8;              // elements at data_size = 8 B
#pragma unroll 1
    for (int g = 0; g < 4; ++g) {
      unsigned long long ga =
          (unsigned long long)(uintptr_t)(WhTd + (size_t)(g * H_ + n0) * H_);
      v4u g0v;
      g0v[0] = 1u;                                   // count = 1 (valid descriptor)
      g0v[1] = lds_base + (unsigned)g * chunkB;      // lds_addr
      g0v[2] = (unsigned)(ga & 0xffffffffu);         // global_addr[31:0]
      g0v[3] = (unsigned)((ga >> 32) & 0x1ffffffu) | (2u << 30);  // addr[56:32] | type=2
      v8i g1v;
      g1v[0] = (int)(3u << 16);                      // workgroup_mask=0, data_size=8B
      g1v[1] = (int)((chunkE & 0xffffu) << 16);      // tensor_dim0[15:0]
      g1v[2] = (int)((chunkE >> 16) | (1u << 16));   // tensor_dim0[31:16] | tensor_dim1=1
      g1v[3] = (int)((chunkE & 0xffffu) << 16);      // tile_dim0 = chunkE
      g1v[4] = (int)(1u | (1u << 16));               // tile_dim1 = 1, tile_dim2 = 1
      g1v[5] = (int)chunkE;                          // tensor_dim0_stride
      g1v[6] = 0;
      g1v[7] = 0;
      v4i z4 = {0, 0, 0, 0};
#if defined(__clang_major__) && (__clang_major__ >= 23)
      v8i z8 = {0, 0, 0, 0, 0, 0, 0, 0};
      __builtin_amdgcn_tensor_load_to_lds(g0v, g1v, z4, z4, z8, 0);
#else
      __builtin_amdgcn_tensor_load_to_lds(g0v, g1v, z4, z4, 0);
#endif
    }
    __builtin_amdgcn_s_wait_tensorcnt(0);
  }
#else
  {  // fallback: cooperative copy (8192 B128 rows -> LDS)
    const uint4* dummy;
    uint4* dst = reinterpret_cast<uint4*>(sh_wh);
    for (int i = threadIdx.x; i < 4 * NCOLS * (H_ / 8); i += 256) {
      int g   = i / (NCOLS * (H_ / 8));
      int r   = i % (NCOLS * (H_ / 8));
      int col = r / (H_ / 8);
      int k   = r % (H_ / 8);
      dst[i] = reinterpret_cast<const uint4*>(
                   WhTd + (size_t)(g * H_ + n0 + col) * H_)[k];
    }
    (void)dummy;
  }
#endif
  __syncthreads();

  // Hoist bias (constant across time) into registers.
  float bv[2][4];
#pragma unroll
  for (int n = 0; n < 2; ++n)
#pragma unroll
    for (int g = 0; g < 4; ++g) bv[n][g] = bias[g * H_ + n0 + n * 16 + lr];

  v8f cst[2];
#pragma unroll
  for (int n = 0; n < 2; ++n)
#pragma unroll
    for (int r = 0; r < 8; ++r) cst[n][r] = 0.0f;

  for (int t = 0; t < T_; ++t) {
    const int tx = dir ? (T_ - 1 - t) : t;
    const unsigned short* Xrow = X + ((size_t)(seq * T_ + tx) * B_) * KX;
    const unsigned short* hr = Hbuf + (size_t)(t & 1) * hstride + (size_t)rec * B_ * H_;
    unsigned short* hw = Hbuf + (size_t)((t + 1) & 1) * hstride + (size_t)rec * B_ * H_;
    unsigned short* ho = Hout + (((size_t)(seq * 2 + dir) * T_ + tx) * B_) * H_;

#pragma unroll 1
    for (int n = 0; n < 2; ++n) {
      const int ncol = n0 + n * 16;
      v8f acc[4];
#pragma unroll
      for (int g = 0; g < 4; ++g)
#pragma unroll
        for (int r = 0; r < 8; ++r) acc[g][r] = bv[n][g];

      // x_t @ Wx  (K = 320 padded); B fragments stream from L2-resident WxT
#pragma unroll 1
      for (int k0 = 0; k0 < KX; k0 += 32) {
        v16bf Am = load_frag(Xrow + (size_t)(m0 + lr) * KX + k0 + kh * 8);
#pragma unroll
        for (int g = 0; g < 4; ++g) {
          v16bf Bm = load_frag(WxTd + (size_t)(g * H_ + ncol + lr) * KX + k0 + kh * 8);
          acc[g] = __builtin_amdgcn_wmma_f32_16x16x32_bf16(
              false, Am, false, Bm, (short)0, acc[g], false, false);
        }
      }
      // h_{t-1} @ Wh  (K = 512); B fragments from the LDS-staged Wh slice
#pragma unroll 1
      for (int k0 = 0; k0 < H_; k0 += 32) {
        v16bf Am = load_frag(hr + (size_t)(m0 + lr) * H_ + k0 + kh * 8);
#pragma unroll
        for (int g = 0; g < 4; ++g) {
          v16bf Bm = load_frag(&sh_wh[(size_t)(g * NCOLS + n * 16 + lr) * H_ + k0 + kh * 8]);
          acc[g] = __builtin_amdgcn_wmma_f32_16x16x32_bf16(
              false, Am, false, Bm, (short)0, acc[g], false, false);
        }
      }

      // Gate math: all four gate tiles are element-aligned in the C layout.
#pragma unroll
      for (int r = 0; r < 8; ++r) {
        float iv = acc[0][r], jv = acc[1][r], fv = acc[2][r], ov = acc[3][r];
        float c2 = cst[n][r] * sigm(fv + 1.0f) + sigm(iv) * ftanh(jv);
        float hv = ftanh(c2) * sigm(ov);
        cst[n][r] = c2;
        const int row = m0 + r + 8 * kh;
        const int col = ncol + lr;
        unsigned short hb = f2bf(hv);
        hw[(size_t)row * H_ + col] = hb;
        ho[(size_t)row * H_ + col] = hb;
      }
    }

    if (t + 1 < T_) {  // hint next timestep's x rows toward L2/WGP$
      const int txn = dir ? (T_ - 2 - t) : (t + 1);
      __builtin_prefetch(X + ((size_t)(seq * T_ + txn) * B_) * KX + (size_t)(m0 + lr) * KX, 0, 1);
    }

    // Cross-workgroup step barrier for this recurrence (release/acquire counter).
    __threadfence();
    __syncthreads();
    if (threadIdx.x == 0) {
      __hip_atomic_fetch_add(syncc + rec, 1u, __ATOMIC_RELEASE, __HIP_MEMORY_SCOPE_AGENT);
      const unsigned target = (unsigned)WGS_PER_REC * (unsigned)(t + 1);
      while (__hip_atomic_load(syncc + rec, __ATOMIC_ACQUIRE, __HIP_MEMORY_SCOPE_AGENT) < target) {
        __builtin_amdgcn_s_sleep(1);
      }
    }
    __syncthreads();
  }
}

// ---------------- attention kernels ----------------

__global__ void k_qmean(const unsigned short* __restrict__ Hout,
                        float* __restrict__ qm, float* __restrict__ out) {
  int idx = blockIdx.x * blockDim.x + threadIdx.x;   // B * 1024
  if (idx >= B_ * 1024) return;
  int b = idx >> 10, h2 = idx & 1023;
  int dir = h2 >> 9, h = h2 & 511;
  const unsigned short* base =
      Hout + ((size_t)(0 * 2 + dir) * T_ * B_) * H_ + (size_t)b * H_ + h;
  float s = 0.0f;
  for (int t = 0; t < T_; ++t) s += bf2f(base[(size_t)t * B_ * H_]);
  s *= (1.0f / T_);
  qm[idx] = s;
  out[idx] = s;                        // row 0: eq_pos
  out[2 * B_ * 1024 + idx] = s;        // row 2: eq_neg
}

__global__ void k_uqm(const float* __restrict__ U, const float* __restrict__ qm,
                      float* __restrict__ v) {
  int idx = blockIdx.x * blockDim.x + threadIdx.x;   // B * 1024
  if (idx >= B_ * 1024) return;
  int b = idx >> 10, h = idx & 1023;
  const float* Ur = U + (size_t)h * 1024;
  const float* q  = qm + (size_t)b * 1024;
  float s = 0.0f;
  for (int k = 0; k < 1024; ++k) s += Ur[k] * q[k];
  v[idx] = s;
}

__global__ void __launch_bounds__(256)
k_attn(const unsigned short* __restrict__ Hout, const float* __restrict__ v,
       float* __restrict__ out) {
  const int aidx = blockIdx.x >> 7;       // 0 = a_pos, 1 = a_neg
  const int b    = blockIdx.x & 127;
  const int seq  = 1 + aidx;
  __shared__ float s_v[1024];
  __shared__ float s_al[256];
  __shared__ float s_red[256];
  const int tid = threadIdx.x;
  for (int h = tid; h < 1024; h += 256) s_v[h] = v[(size_t)b * 1024 + h];
  __syncthreads();

  const unsigned short* fw = Hout + (((size_t)(seq * 2 + 0) * T_ + tid) * B_ + b) * H_;
  const unsigned short* bw = Hout + (((size_t)(seq * 2 + 1) * T_ + tid) * B_ + b) * H_;
  float sc = 0.0f;
  for (int h = 0; h < 512; ++h) sc += bf2f(fw[h]) * s_v[h];
  for (int h = 0; h < 512; ++h) sc += bf2f(bw[h]) * s_v[512 + h];

  s_red[tid] = sc; __syncthreads();
  for (int s = 128; s > 0; s >>= 1) {
    if (tid < s) s_red[tid] = fmaxf(s_red[tid], s_red[tid + s]);
    __syncthreads();
  }
  float mx = s_red[0]; __syncthreads();
  float e = __expf(sc - mx);
  s_red[tid] = e; __syncthreads();
  for (int s = 128; s > 0; s >>= 1) {
    if (tid < s) s_red[tid] += s_red[tid + s];
    __syncthreads();
  }
  float inv = 1.0f / s_red[0];
  s_al[tid] = e * inv; __syncthreads();

  for (int h2 = tid; h2 < 1024; h2 += 256) {
    int dir = h2 >> 9, h = h2 & 511;
    const unsigned short* base =
        Hout + ((size_t)(seq * 2 + dir) * T_ * B_) * H_ + (size_t)b * H_ + h;
    float acc = 0.0f;
    for (int t = 0; t < T_; ++t) acc += s_al[t] * bf2f(base[(size_t)t * B_ * H_]);
    out[(size_t)(1 + 2 * aidx) * B_ * 1024 + (size_t)b * 1024 + h2] = acc;
  }
}

// ---------------- launcher ----------------

extern "C" void kernel_launch(void* const* d_in, const int* in_sizes, int n_in,
                              void* d_out, int out_size, void* d_ws, size_t ws_size,
                              hipStream_t stream) {
  const float* q  = (const float*)d_in[0];
  const float* a  = (const float*)d_in[1];
  const float* an = (const float*)d_in[2];
  const float* Wf = (const float*)d_in[3];
  const float* bf = (const float*)d_in[4];
  const float* Wb = (const float*)d_in[5];
  const float* bb = (const float*)d_in[6];
  const float* U  = (const float*)d_in[7];
  float* out = (float*)d_out;
  char* ws = (char*)d_ws;

  const size_t o_sync = 0;
  const size_t o_hbuf = 256;
  const size_t o_x    = o_hbuf + (size_t)2 * NREC * B_ * H_ * 2;   // double-buffered h
  const size_t o_wxT  = o_x    + (size_t)3 * T_ * B_ * KX * 2;
  const size_t o_whT  = o_wxT  + (size_t)2 * G4 * KX * 2;
  const size_t o_hout = o_whT  + (size_t)2 * G4 * H_ * 2;
  const size_t o_qm   = o_hout + (size_t)3 * 2 * T_ * B_ * H_ * 2;
  const size_t o_v    = o_qm   + (size_t)B_ * 1024 * 4;

  unsigned*       syncc = (unsigned*)(ws + o_sync);
  unsigned short* Hbuf  = (unsigned short*)(ws + o_hbuf);
  unsigned short* X     = (unsigned short*)(ws + o_x);
  unsigned short* WxT   = (unsigned short*)(ws + o_wxT);
  unsigned short* WhT   = (unsigned short*)(ws + o_whT);
  unsigned short* Hout  = (unsigned short*)(ws + o_hout);
  float*          qm    = (float*)(ws + o_qm);
  float*          vv    = (float*)(ws + o_v);

  // Zero the step-barrier counters and the initial h double-buffer.
  hipMemsetAsync(ws, 0, o_x, stream);

  {
    long total = (long)3 * T_ * B_ * KX;
    k_convert_x<<<(unsigned)((total + 255) / 256), 256, 0, stream>>>(q, a, an, X);
  }
  {
    long total = (long)2 * G4 * KX;
    k_build_wxT<<<(unsigned)((total + 255) / 256), 256, 0, stream>>>(Wf, Wb, WxT);
  }
  {
    long total = (long)2 * G4 * H_;
    k_build_whT<<<(unsigned)((total + 255) / 256), 256, 0, stream>>>(Wf, Wb, WhT);
  }

  (void)hipFuncSetAttribute((const void*)k_lstm,
                            hipFuncAttributeMaxDynamicSharedMemorySize, LDS_BYTES);
  k_lstm<<<NREC * WGS_PER_REC, 256, LDS_BYTES, stream>>>(X, WxT, WhT, bf, bb,
                                                         Hbuf, Hout, syncc);

  k_qmean<<<(B_ * 1024) / 256, 256, 0, stream>>>(Hout, qm, out);
  k_uqm<<<(B_ * 1024) / 256, 256, 0, stream>>>(U, qm, vv);
  k_attn<<<2 * B_, 256, 0, stream>>>(Hout, vv, out);
}
